// LocalTokenMerger_32066225832620
// MI455X (gfx1250) — compile-verified, matrix operand order checked by reading, not run on hardware
//
#include <hip/hip_runtime.h>
#include <hip/hip_bf16.h>
#include <cstdint>

typedef __attribute__((ext_vector_type(16))) _Float16 v16h;
typedef __attribute__((ext_vector_type(8)))  float    v8f;

#define B_      8
#define T_      8192
#define D_      1024
#define GD_     64
#define WSZ_    16
#define NWIN_   512                 // T_/WSZ_
#define NWINDOWS_ (B_*NWIN_)        // 4096
#define TARGET_ 6144
#define KEEP_   12                  // 16 - budget, budget==4 for this shape
#define BUDGET_ 4

// ---- workspace layout (bytes) ----
// W1 fragments : 32 ksteps * 4 ntiles * 32 lanes * 8 dwords = 131072 B
// W2 fragments :  2 ksteps * 4 ntiles * 32 lanes * 8 dwords =   8192 B
// merge masks  : 4096 * 4                                   =  16384 B
// lens (int)   : 8 * 6144 * 4                               = 196608 B
#define W1F_OFF   0
#define W2F_OFF   131072
#define MASK_OFF  139264
#define LENS_OFF  155648

union AFrag { v16h v; _Float16 h[16]; };
union BFrag { v16h v; uint4 q[2]; };

// K offset inside a 32-wide k-step for VGPR slot i, per CDNA5 16-bit A/B layouts
__device__ __forceinline__ int kofs(int i, int half) {
    return 2 * (i & 3) + ((i >> 2) << 4) + (half << 3);
}

// ---------------------------------------------------------------------------
// Kernel 0: pack W1/W2 (f32 row-major KxN) into f16 WMMA B-fragment order.
// Fragment (ks, nt): lane l holds column n = nt*16 + (l&15); dword i packs
// f16(W[k0][n]), f16(W[k0+1][n]) with k0 = ks*32 + kofs(i, l>>4).
// ---------------------------------------------------------------------------
__global__ __launch_bounds__(256)
void prep_wfrag(const float* __restrict__ W1, const float* __restrict__ W2,
                uint32_t* __restrict__ w1f, uint32_t* __restrict__ w2f) {
    int tid = blockIdx.x * blockDim.x + threadIdx.x;
    if (tid < 32768) {
        int i = tid & 7, lane = (tid >> 3) & 31, nt = (tid >> 8) & 3, ks = tid >> 10;
        int n  = nt * 16 + (lane & 15);
        int k0 = ks * 32 + kofs(i, lane >> 4);
        union { _Float16 h[2]; uint32_t u; } pk;
        pk.h[0] = (_Float16)W1[k0 * GD_ + n];
        pk.h[1] = (_Float16)W1[(k0 + 1) * GD_ + n];
        w1f[tid] = pk.u;
    } else if (tid < 32768 + 2048) {
        int t2 = tid - 32768;
        int i = t2 & 7, lane = (t2 >> 3) & 31, nt = (t2 >> 8) & 3, ks = (t2 >> 10) & 1;
        int n  = nt * 16 + (lane & 15);
        int k0 = ks * 32 + kofs(i, lane >> 4);
        union { _Float16 h[2]; uint32_t u; } pk;
        pk.h[0] = (_Float16)W2[k0 * GD_ + n];
        pk.h[1] = (_Float16)W2[(k0 + 1) * GD_ + n];
        w2f[t2] = pk.u;
    }
}

// ---------------------------------------------------------------------------
// Kernel 1: per-window gate GEMMs (WMMA f16), normalize, sims, greedy select.
// One wave32 per 16-token window; 4 waves per block.
// ---------------------------------------------------------------------------
__global__ __launch_bounds__(128)
void gate_select(const float* __restrict__ z,
                 const uint32_t* __restrict__ w1f,
                 const uint32_t* __restrict__ w2f,
                 uint32_t* __restrict__ masks) {
    __shared__ _Float16 sH[4][16][64];   // relu(z@W1) staged for 2nd GEMM
    __shared__ float    sG[4][16][64];   // gate output
    __shared__ float    sN[4][16];       // row norms
    __shared__ float    sS[4][16];       // 15 edge sims

    const int lane = threadIdx.x & 31;
    const int wv   = threadIdx.x >> 5;
    const int win  = blockIdx.x * 4 + wv;       // flat window id (b*512 + w)
    const int m    = lane & 15;
    const int half = lane >> 4;

    const float* zrow = z + ((size_t)win * WSZ_ + m) * (size_t)D_;

    const v8f zero = {0.f, 0.f, 0.f, 0.f, 0.f, 0.f, 0.f, 0.f};
    v8f acc[4];
#pragma unroll
    for (int nt = 0; nt < 4; ++nt) acc[nt] = zero;

    // GEMM1: (16x1024) @ (1024x64): 32 k-steps of 32, 4 n-tiles
    for (int ks = 0; ks < 32; ++ks) {
        if (ks + 4 < 32) __builtin_prefetch(zrow + (ks + 4) * 32, 0, 0);
        AFrag a;
#pragma unroll
        for (int i = 0; i < 8; ++i) {
            int k0 = ks * 32 + kofs(i, half);
            float2 f = *(const float2*)(zrow + k0);
            a.h[2 * i]     = (_Float16)f.x;
            a.h[2 * i + 1] = (_Float16)f.y;
        }
#pragma unroll
        for (int nt = 0; nt < 4; ++nt) {
            BFrag bf;
            const uint4* p = (const uint4*)(w1f + ((size_t)(ks * 4 + nt) * 32 + lane) * 8);
            bf.q[0] = p[0];
            bf.q[1] = p[1];
            acc[nt] = __builtin_amdgcn_wmma_f32_16x16x32_f16(
                false, a.v, false, bf.v, (short)0, acc[nt], false, false);
        }
    }

    // ReLU, stage H (f16) to LDS in row-major for the 2nd GEMM's A fragments
#pragma unroll
    for (int nt = 0; nt < 4; ++nt)
#pragma unroll
        for (int r = 0; r < 8; ++r)
            sH[wv][r + (half << 3)][nt * 16 + (lane & 15)] =
                (_Float16)fmaxf(acc[nt][r], 0.0f);

    // GEMM2: (16x64) @ (64x64): 2 k-steps, 4 n-tiles
    v8f g[4];
#pragma unroll
    for (int nt = 0; nt < 4; ++nt) g[nt] = zero;
#pragma unroll
    for (int ks = 0; ks < 2; ++ks) {
        AFrag a;
#pragma unroll
        for (int i = 0; i < 8; ++i) {
            int k0 = ks * 32 + kofs(i, half);
            a.h[2 * i]     = sH[wv][m][k0];
            a.h[2 * i + 1] = sH[wv][m][k0 + 1];
        }
#pragma unroll
        for (int nt = 0; nt < 4; ++nt) {
            BFrag bf;
            const uint4* p = (const uint4*)(w2f + ((size_t)(ks * 4 + nt) * 32 + lane) * 8);
            bf.q[0] = p[0];
            bf.q[1] = p[1];
            g[nt] = __builtin_amdgcn_wmma_f32_16x16x32_f16(
                false, a.v, false, bf.v, (short)0, g[nt], false, false);
        }
    }

    // Scatter G to LDS (f32): C/D layout is (M = r + 8*half, N = nt*16 + lane&15)
#pragma unroll
    for (int nt = 0; nt < 4; ++nt)
#pragma unroll
        for (int r = 0; r < 8; ++r)
            sG[wv][r + (half << 3)][nt * 16 + (lane & 15)] = g[nt][r];

    // Row norms (lanes 0..15), then adjacent sims (lanes 0..14).
    // Same-wave LDS ops are in-order -> no barrier needed (private slice).
    if (lane < 16) {
        float s = 0.f;
        for (int j = 0; j < GD_; ++j) { float v = sG[wv][lane][j]; s += v * v; }
        sN[wv][lane] = sqrtf(s) + 1e-8f;
    }
    if (lane < 15) {
        float d = 0.f;
        for (int j = 0; j < GD_; ++j) d += sG[wv][lane][j] * sG[wv][lane + 1][j];
        sS[wv][lane] = d / (sN[wv][lane] * sN[wv][lane + 1]);
    }

    // Greedy matching, stable-descending order (first index wins ties),
    // budget = 4. Matches reference _choose_pairs exactly.
    if (lane == 0) {
        float s[15];
        for (int e = 0; e < 15; ++e) s[e] = sS[wv][e];
        unsigned used = 0, proc = 0, mr = 0;
        int picked = 0;
        for (int it = 0; it < 15 && picked < BUDGET_; ++it) {
            int best = -1; float bv = -3.0e38f;
            for (int e = 0; e < 15; ++e)
                if (!((proc >> e) & 1) && s[e] > bv) { bv = s[e]; best = e; }
            if (best < 0) break;
            proc |= 1u << best;
            if (!((used >> best) & 1) && !((used >> (best + 1)) & 1)) {
                used |= 3u << best;       // mark best, best+1
                mr   |= 1u << (best + 1); // merge_right at right token
                ++picked;
            }
        }
        masks[win] = mr;
    }
}

// ---------------------------------------------------------------------------
// Kernel 2: weighted merge + compaction. One 256-thread block per window;
// 12 output rows of 1024 f32 each (256 float4 columns = 1 per thread).
// ---------------------------------------------------------------------------
__global__ __launch_bounds__(256)
void merge_compact(const float* __restrict__ z,
                   const int* __restrict__ lens,
                   const uint32_t* __restrict__ masks,
                   float* __restrict__ out_z,
                   float* __restrict__ out_lens_f,
                   int* __restrict__ lens_i) {
    const int win = blockIdx.x;
    const uint32_t mask = masks[win];
    const float* zwin = z + (size_t)win * WSZ_ * D_;
    const int*   lw   = lens + win * WSZ_;
    float* oz = out_z + (size_t)win * KEEP_ * D_;   // b*6144 + w*12 == win*12

    int kept[KEEP_];
#pragma unroll
    for (int k = 0; k < KEEP_; ++k) kept[k] = WSZ_ - 1;
    int kc = 0;
    for (int t = 0; t < WSZ_; ++t)
        if (!((mask >> t) & 1)) { if (kc < KEEP_) kept[kc] = t; ++kc; }

    const int c = threadIdx.x;   // float4 column
    for (int k = 0; k < KEEP_; ++k) {
        int t = kept[k];
        bool absorb = (t < WSZ_ - 1) && ((mask >> (t + 1)) & 1);
        float wa = (float)lw[t];
        float4 va = ((const float4*)(zwin + (size_t)t * D_))[c];
        float4 o;
        if (absorb) {
            float wb  = (float)lw[t + 1];
            float inv = 1.0f / (wa + wb);
            float4 vb = ((const float4*)(zwin + (size_t)(t + 1) * D_))[c];
            o.x = (va.x * wa + vb.x * wb) * inv;
            o.y = (va.y * wa + vb.y * wb) * inv;
            o.z = (va.z * wa + vb.z * wb) * inv;
            o.w = (va.w * wa + vb.w * wb) * inv;
        } else {
            o = va;
        }
        ((float4*)(oz + (size_t)k * D_))[c] = o;
        if (threadIdx.x == 0) {
            int ln = absorb ? (lw[t] + lw[t + 1]) : lw[t];
            lens_i[win * KEEP_ + k]    = ln;
            out_lens_f[win * KEEP_ + k] = (float)ln;
        }
    }
}

// ---------------------------------------------------------------------------
// Kernel 3: per-batch exclusive prefix sum of lens -> starts. One block/batch.
// ---------------------------------------------------------------------------
__global__ __launch_bounds__(256)
void scan_starts(const int* __restrict__ lens_i, float* __restrict__ out_starts) {
    __shared__ int sb[256];
    __shared__ int carry;
    const int b = blockIdx.x, tid = threadIdx.x;
    if (tid == 0) carry = 0;
    __syncthreads();
    for (int chunk = 0; chunk < TARGET_ / 256; ++chunk) {
        int idx = b * TARGET_ + chunk * 256 + tid;
        int v = lens_i[idx];
        sb[tid] = v;
        __syncthreads();
        for (int off = 1; off < 256; off <<= 1) {
            int t = (tid >= off) ? sb[tid - off] : 0;
            __syncthreads();
            sb[tid] += t;
            __syncthreads();
        }
        int incl = sb[tid];
        int base = carry;
        out_starts[idx] = (float)(base + incl - v);   // exclusive
        __syncthreads();
        if (tid == 255) carry = base + incl;
        __syncthreads();
    }
}

// ---------------------------------------------------------------------------
extern "C" void kernel_launch(void* const* d_in, const int* in_sizes, int n_in,
                              void* d_out, int out_size, void* d_ws, size_t ws_size,
                              hipStream_t stream) {
    const float* z  = (const float*)d_in[0];
    const int*   tl = (const int*)d_in[1];
    const float* W1 = (const float*)d_in[2];
    const float* W2 = (const float*)d_in[3];
    // d_in[4] = target_len (6144): fixed for this shape (budget=4, keep=12)

    char* ws = (char*)d_ws;
    uint32_t* w1f    = (uint32_t*)(ws + W1F_OFF);
    uint32_t* w2f    = (uint32_t*)(ws + W2F_OFF);
    uint32_t* masks  = (uint32_t*)(ws + MASK_OFF);
    int*      lens_i = (int*)(ws + LENS_OFF);

    float* out        = (float*)d_out;
    float* out_z      = out;                                   // 8*6144*1024
    float* out_lens   = out + (size_t)B_ * TARGET_ * D_;       // 8*6144
    float* out_starts = out_lens + (size_t)B_ * TARGET_;       // 8*6144

    prep_wfrag<<<(32768 + 2048 + 255) / 256, 256, 0, stream>>>(W1, W2, w1f, w2f);
    gate_select<<<NWINDOWS_ / 4, 128, 0, stream>>>(z, w1f, w2f, masks);
    merge_compact<<<NWINDOWS_, 256, 0, stream>>>(z, tl, masks, out_z, out_lens, lens_i);
    scan_starts<<<B_, 256, 0, stream>>>(lens_i, out_starts);
}
